// LSTM_3_68693706932502
// MI455X (gfx1250) — compile-verified, matrix operand order checked by reading, not run on hardware
//
#include <hip/hip_runtime.h>

// Problem dims (fixed by the reference)
constexpr int kB = 32, kS = 96, kT = 96, kE = 256, kH = 512, kV = 32000;

typedef __attribute__((ext_vector_type(16))) __bf16 v16bf;
typedef __attribute__((ext_vector_type(8)))  __bf16 v8bf;
typedef __attribute__((ext_vector_type(8)))  float  v8f;

__device__ __forceinline__ v8f wmma_bf16(v16bf a, v16bf b, v8f c) {
  // (neg_a, A, neg_b, B, c_mod, C, reuse_a, reuse_b)
  return __builtin_amdgcn_wmma_f32_16x16x32_bf16(false, a, false, b, (short)0, c,
                                                 false, false);
}

// Load one 16-lane-group operand fragment. Per CDNA5 ISA 16-bit A/B layout:
// lane group g holds K = {8g..8g+7, 16+8g..23+8g}; p must point at (row base + 8g),
// 16-byte aligned.
__device__ __forceinline__ v16bf load_frag(const __bf16* p) {
  v8bf lo = *(const v8bf*)p;
  v8bf hi = *(const v8bf*)(p + 16);
  return __builtin_shufflevector(lo, hi, 0, 1, 2, 3, 4, 5, 6, 7,
                                 8, 9, 10, 11, 12, 13, 14, 15);
}

// ---------------------------------------------------------------------------
// fp32 -> bf16 conversion
__global__ void cvt_kernel(const float* __restrict__ s, __bf16* __restrict__ d,
                           long long n) {
  long long i = (long long)blockIdx.x * 256 + threadIdx.x;
  if (i < n) d[i] = (__bf16)s[i];
}

// Embedding gather: out[t][b][e] (bf16), seq is [B, T]
__global__ void embed_kernel(const int* __restrict__ seq,
                             const float* __restrict__ emb,
                             __bf16* __restrict__ out, int T_) {
  long long i = (long long)blockIdx.x * 256 + threadIdx.x;
  long long total = (long long)T_ * kB * kE;
  if (i >= total) return;
  int e = (int)(i % kE);
  long long tb = i / kE;
  int b = (int)(tb % kB);
  int t = (int)(tb / kB);
  int tok = seq[b * T_ + t];
  out[i] = (__bf16)emb[(long long)tok * kE + e];
}

// ---------------------------------------------------------------------------
// WMMA GEMM: out = A[M,K](bf16,row-major) x Bw[N,K]^T(bf16) + bias
// Block tile 128x256, 512 threads / 16 waves; wave = 32x64 quad (2x4 tiles of
// 16x16 -> 8 WMMAs per 32-wide k-chunk). Double-buffered LDS, one barrier per
// k-chunk: stage k+1 global loads are in flight across stage-k compute.
// Output scatter: m decomposed as m = t*mdiv + b; off = t*sq + b*sr + n.
__global__ __launch_bounds__(512) void gemm_wmma(
    const __bf16* __restrict__ A, const __bf16* __restrict__ Bw,
    const float* __restrict__ bias, float* __restrict__ out,
    int K, int mdiv, long long sq, long long sr) {
  __shared__ __bf16 As[2][128][40];  // 80B row stride -> conflict-free b128
  __shared__ __bf16 Bs[2][256][40];
  const int n0 = blockIdx.x * 256;
  const int m0 = blockIdx.y * 128;
  const int tid = threadIdx.x;
  const int lane = tid & 31, wave = tid >> 5;
  const int wm = wave >> 2, wn = wave & 3;   // 4x4 wave grid
  const int g = lane >> 4, r = lane & 15;
  const int arow = tid >> 2, acol = (tid & 3) * 8;   // A staging: 128 x 32
  const int brow = tid >> 1, bcol = (tid & 1) * 16;  // B staging: 256 x 32

  const __bf16* gA = A + (size_t)(m0 + arow) * K + acol;
  const __bf16* gB = Bw + (size_t)(n0 + brow) * K + bcol;

  v8f acc[2][4] = {};

  // Prologue: stage 0 -> buffer 0
  {
    v8bf ra = *(const v8bf*)gA;
    v8bf rb0 = *(const v8bf*)gB;
    v8bf rb1 = *(const v8bf*)(gB + 8);
    *(v8bf*)&As[0][arow][acol] = ra;
    *(v8bf*)&Bs[0][brow][bcol] = rb0;
    *(v8bf*)&Bs[0][brow][bcol + 8] = rb1;
  }
  __syncthreads();

  int cur = 0;
  for (int k0 = 32; k0 < K; k0 += 32, cur ^= 1) {
    // Issue next-stage global loads (overlap with WMMA below)
    v8bf ra = *(const v8bf*)(gA + k0);
    v8bf rb0 = *(const v8bf*)(gB + k0);
    v8bf rb1 = *(const v8bf*)(gB + k0 + 8);
    // Compute current stage from buf[cur]
    v16bf a0 = load_frag(&As[cur][wm * 32 + r][8 * g]);
    v16bf a1 = load_frag(&As[cur][wm * 32 + 16 + r][8 * g]);
#pragma unroll
    for (int j = 0; j < 4; ++j) {
      v16bf bf = load_frag(&Bs[cur][(wn * 4 + j) * 16 + r][8 * g]);
      acc[0][j] = wmma_bf16(a0, bf, acc[0][j]);
      acc[1][j] = wmma_bf16(a1, bf, acc[1][j]);
    }
    // Stage next buffer (no conflict with concurrent reads of buf[cur])
    *(v8bf*)&As[cur ^ 1][arow][acol] = ra;
    *(v8bf*)&Bs[cur ^ 1][brow][bcol] = rb0;
    *(v8bf*)&Bs[cur ^ 1][brow][bcol + 8] = rb1;
    __syncthreads();
  }
  // Epilogue: last stage
  {
    v16bf a0 = load_frag(&As[cur][wm * 32 + r][8 * g]);
    v16bf a1 = load_frag(&As[cur][wm * 32 + 16 + r][8 * g]);
#pragma unroll
    for (int j = 0; j < 4; ++j) {
      v16bf bf = load_frag(&Bs[cur][(wn * 4 + j) * 16 + r][8 * g]);
      acc[0][j] = wmma_bf16(a0, bf, acc[0][j]);
      acc[1][j] = wmma_bf16(a1, bf, acc[1][j]);
    }
  }

#pragma unroll
  for (int rt = 0; rt < 2; ++rt)
#pragma unroll
    for (int j = 0; j < 4; ++j) {
      int n = n0 + (wn * 4 + j) * 16 + r;
      float bv = bias ? bias[n] : 0.f;
#pragma unroll
      for (int i = 0; i < 8; ++i) {
        int m = m0 + wm * 32 + rt * 16 + 8 * g + i;
        int tt = m / mdiv;
        int bb = m - tt * mdiv;
        out[(long long)tt * sq + (long long)bb * sr + n] = acc[rt][j][i] + bv;
      }
    }
}

// ---------------------------------------------------------------------------
// Persistent LSTM recurrence: one workgroup (512 thr / 16 waves) runs all S_
// steps. g = xp[t] + h @ Whh^T (WMMA bf16), then gates in registers.
// Wave w owns H-columns [w*32, w*32+32) split in two 16-col groups; it computes
// the i/f/g/o tiles for those columns so the elementwise gate math needs no
// cross-wave exchange. h kept in LDS (bf16), c kept in accumulator-layout regs.
__global__ __launch_bounds__(512) void lstm_seq(
    const float* __restrict__ xp,      // [S_, 32, 2048]
    const __bf16* __restrict__ Whh,    // [2048, 512]
    const float* __restrict__ h0,      // [32,512] or null
    const float* __restrict__ c0,      // [32,512] or null
    __bf16* __restrict__ outBf,        // [S_,32,ldo] (+co) or null
    float* __restrict__ outF,          // same indexing, or null
    float* __restrict__ hT, float* __restrict__ cT,  // [32,ldT] (+co) or null
    int S_, int ldo, int co, int ldT, int reverse) {
  __shared__ __bf16 hbuf[32][520];  // padded: 1040B row stride
  const int tid = threadIdx.x, lane = tid & 31, wave = tid >> 5;  // wave 0..15
  const int g = lane >> 4, r15 = lane & 15;

  for (int i = tid; i < 32 * 512; i += 512) {
    int row = i >> 9, col = i & 511;
    hbuf[row][col] = (__bf16)(h0 ? h0[i] : 0.f);
  }
  v8f cacc[2][2] = {};  // [colgroup][rowblock]
  if (c0) {
#pragma unroll
    for (int cg = 0; cg < 2; ++cg)
#pragma unroll
      for (int rb = 0; rb < 2; ++rb)
#pragma unroll
        for (int i = 0; i < 8; ++i) {
          int row = rb * 16 + 8 * g + i;
          int col = wave * 32 + cg * 16 + r15;
          cacc[cg][rb][i] = c0[row * 512 + col];
        }
  }
  __syncthreads();

  for (int t = 0; t < S_; ++t) {
    const int tr = reverse ? (S_ - 1 - t) : t;
    const float* xpt = xp + (size_t)tr * 32 * 2048;
    v8f hnew[2][2];
#pragma unroll
    for (int cg = 0; cg < 2; ++cg) {
      const int jc = wave * 32 + cg * 16;
      v8f acc[4][2] = {};
      for (int kc = 0; kc < 512; kc += 32) {
        v16bf a0 = load_frag(&hbuf[r15][kc + 8 * g]);
        v16bf a1 = load_frag(&hbuf[16 + r15][kc + 8 * g]);
#pragma unroll
        for (int gi = 0; gi < 4; ++gi) {
          const __bf16* bp =
              Whh + (size_t)(gi * 512 + jc + r15) * 512 + kc + 8 * g;
          v16bf bfrag = load_frag(bp);
          acc[gi][0] = wmma_bf16(a0, bfrag, acc[gi][0]);
          acc[gi][1] = wmma_bf16(a1, bfrag, acc[gi][1]);
        }
      }
#pragma unroll
      for (int rb = 0; rb < 2; ++rb)
#pragma unroll
        for (int i = 0; i < 8; ++i) {
          int row = rb * 16 + 8 * g + i;
          const float* xr = xpt + row * 2048 + jc + r15;
          float iv = 1.f / (1.f + __expf(-(acc[0][rb][i] + xr[0])));
          float fv = 1.f / (1.f + __expf(-(acc[1][rb][i] + xr[512])));
          float gv = tanhf(acc[2][rb][i] + xr[1024]);
          float ov = 1.f / (1.f + __expf(-(acc[3][rb][i] + xr[1536])));
          float cv = fv * cacc[cg][rb][i] + iv * gv;
          cacc[cg][rb][i] = cv;
          hnew[cg][rb][i] = ov * tanhf(cv);
        }
    }
    __syncthreads();  // everyone done reading old h
#pragma unroll
    for (int cg = 0; cg < 2; ++cg)
#pragma unroll
      for (int rb = 0; rb < 2; ++rb)
#pragma unroll
        for (int i = 0; i < 8; ++i) {
          int row = rb * 16 + 8 * g + i;
          int col = wave * 32 + cg * 16 + r15;
          float hv = hnew[cg][rb][i];
          hbuf[row][col] = (__bf16)hv;
          size_t oidx = (size_t)tr * 32 * ldo + (size_t)row * ldo + co + col;
          if (outBf) outBf[oidx] = (__bf16)hv;
          if (outF) outF[oidx] = hv;
          if (t == S_ - 1) {
            if (hT) hT[row * ldT + co + col] = hv;
            if (cT) cT[row * ldT + co + col] = cacc[cg][rb][i];
          }
        }
    __syncthreads();
  }
}

// ---------------------------------------------------------------------------
// Small dense projection: out[b][j] = bias[j] + x[b][:1024] . W[j][:1024]
__global__ void proj_kernel(const float* __restrict__ x,
                            const float* __restrict__ W,
                            const float* __restrict__ bias,
                            float* __restrict__ out) {
  int i = blockIdx.x * 256 + threadIdx.x;  // over 32*512
  if (i >= 32 * 512) return;
  int j = i & 511, b = i >> 9;
  const float* xr = x + b * 1024;
  const float* wr = W + j * 1024;
  float s = bias[j];
  for (int k = 0; k < 1024; ++k) s += xr[k] * wr[k];
  out[i] = s;
}

// ---------------------------------------------------------------------------
// Fused attention per (t,b): energy + mask + softmax + context, writes
// combined[t][b][0:512]=y (bf16), [512:1024]=context (bf16).
__global__ __launch_bounds__(128) void attn_kernel(
    const float* __restrict__ y,     // [T,32,512]
    const float* __restrict__ enc,   // [S,32,512]
    const int* __restrict__ src_seq, // [32,S]
    __bf16* __restrict__ comb) {     // [T,32,1024]
  int tb = blockIdx.x;
  int t = tb / kB, b = tb % kB;
  __shared__ float ys[512];
  __shared__ float e[kS];
  __shared__ float red[128];
  int tid = threadIdx.x;
  const float* yr = y + ((size_t)t * kB + b) * 512;
  for (int d = tid; d < 512; d += 128) ys[d] = yr[d];
  __syncthreads();
  float mx = -1e30f;
  for (int s = tid; s < kS; s += 128) {
    const float* er = enc + ((size_t)s * kB + b) * 512;
    float acc = 0.f;
    for (int d = 0; d < 512; ++d) acc += ys[d] * er[d];
    if (src_seq[b * kS + s] == 0) acc = -1e10f;  // PAD mask
    e[s] = acc;
    mx = fmaxf(mx, acc);
  }
  red[tid] = mx;
  __syncthreads();
  for (int off = 64; off > 0; off >>= 1) {
    if (tid < off) red[tid] = fmaxf(red[tid], red[tid + off]);
    __syncthreads();
  }
  mx = red[0];
  __syncthreads();
  float sm = 0.f;
  for (int s = tid; s < kS; s += 128) {
    float v = __expf(e[s] - mx);
    e[s] = v;
    sm += v;
  }
  red[tid] = sm;
  __syncthreads();
  for (int off = 64; off > 0; off >>= 1) {
    if (tid < off) red[tid] += red[tid + off];
    __syncthreads();
  }
  float inv = 1.f / red[0];
  __syncthreads();
  __bf16* cr = comb + ((size_t)t * kB + b) * 1024;
  for (int d = tid; d < 512; d += 128) {
    float acc = 0.f;
    for (int s = 0; s < kS; ++s)
      acc += e[s] * enc[((size_t)s * kB + b) * 512 + d];
    cr[d] = (__bf16)ys[d];
    cr[512 + d] = (__bf16)(acc * inv);
  }
}

// ---------------------------------------------------------------------------
extern "C" void kernel_launch(void* const* d_in, const int* in_sizes, int n_in,
                              void* d_out, int out_size, void* d_ws,
                              size_t ws_size, hipStream_t stream) {
  (void)in_sizes; (void)n_in; (void)out_size; (void)ws_size;
  const int* src_seq = (const int*)d_in[0];
  const int* trg_seq = (const int*)d_in[1];
  const float* src_emb = (const float*)d_in[2];
  const float* trg_emb = (const float*)d_in[3];
  const float* eW0f = (const float*)d_in[4];
  const float* eU0f = (const float*)d_in[5];
  const float* eb0f = (const float*)d_in[6];
  const float* eW0b = (const float*)d_in[7];
  const float* eU0b = (const float*)d_in[8];
  const float* eb0b = (const float*)d_in[9];
  const float* eW1f = (const float*)d_in[10];
  const float* eU1f = (const float*)d_in[11];
  const float* eb1f = (const float*)d_in[12];
  const float* eW1b = (const float*)d_in[13];
  const float* eU1b = (const float*)d_in[14];
  const float* eb1b = (const float*)d_in[15];
  const float* out_W = (const float*)d_in[16];
  const float* out_b = (const float*)d_in[17];
  const float* hpW = (const float*)d_in[18];
  const float* hpb = (const float*)d_in[19];
  const float* cpW = (const float*)d_in[20];
  const float* cpb = (const float*)d_in[21];
  const float* dW0 = (const float*)d_in[22];
  const float* dU0 = (const float*)d_in[23];
  const float* db0 = (const float*)d_in[24];
  const float* dW1 = (const float*)d_in[25];
  const float* dU1 = (const float*)d_in[26];
  const float* db1 = (const float*)d_in[27];
  const float* fcW = (const float*)d_in[28];
  const float* fcb = (const float*)d_in[29];
  float* logits = (float*)d_out;

  char* p = (char*)d_ws;
  auto carve = [&](size_t bytes) {
    char* r = p;
    p += (bytes + 255) & ~(size_t)255;
    return r;
  };
  const size_t M = (size_t)kS * kB;  // 3072 rows for all sequence GEMMs

  __bf16* xsrc = (__bf16*)carve(M * kE * 2);
  __bf16* xtrg = (__bf16*)carve(M * kE * 2);
  float* xp = (float*)carve(M * 4 * kH * 4);
  __bf16* enc0 = (__bf16*)carve(M * 2 * kH * 2);
  __bf16* enc1 = (__bf16*)carve(M * 2 * kH * 2);
  __bf16* dec0 = (__bf16*)carve(M * kH * 2);
  float* yf = (float*)carve(M * kH * 4);
  float* encO = (float*)carve(M * kH * 4);
  __bf16* comb = (__bf16*)carve(M * 2 * kH * 2);
  float* hT0 = (float*)carve(kB * 1024 * 4);
  float* cT0 = (float*)carve(kB * 1024 * 4);
  float* hT1 = (float*)carve(kB * 1024 * 4);
  float* cT1 = (float*)carve(kB * 1024 * 4);
  float* h00 = (float*)carve(kB * 512 * 4);
  float* c00 = (float*)carve(kB * 512 * 4);
  float* h01 = (float*)carve(kB * 512 * 4);
  float* c01 = (float*)carve(kB * 512 * 4);

  auto cvt = [&](const float* s, long long n) {
    __bf16* d = (__bf16*)carve((size_t)n * 2);
    cvt_kernel<<<dim3((unsigned)((n + 255) / 256)), 256, 0, stream>>>(s, d, n);
    return d;
  };
  __bf16* eW0f_b = cvt(eW0f, 2048LL * 256);
  __bf16* eU0f_b = cvt(eU0f, 2048LL * 512);
  __bf16* eW0b_b = cvt(eW0b, 2048LL * 256);
  __bf16* eU0b_b = cvt(eU0b, 2048LL * 512);
  __bf16* eW1f_b = cvt(eW1f, 2048LL * 1024);
  __bf16* eU1f_b = cvt(eU1f, 2048LL * 512);
  __bf16* eW1b_b = cvt(eW1b, 2048LL * 1024);
  __bf16* eU1b_b = cvt(eU1b, 2048LL * 512);
  __bf16* outW_b = cvt(out_W, 512LL * 1024);
  __bf16* dW0_b = cvt(dW0, 2048LL * 256);
  __bf16* dU0_b = cvt(dU0, 2048LL * 512);
  __bf16* dW1_b = cvt(dW1, 2048LL * 512);
  __bf16* dU1_b = cvt(dU1, 2048LL * 512);
  __bf16* fcW_b = cvt(fcW, (long long)kV * 1024);

  auto gemm = [&](const __bf16* A, const __bf16* Bw, const float* bias,
                  float* out, int Mrows, int N, int K, int mdiv, long long sq,
                  long long sr) {
    gemm_wmma<<<dim3(N / 256, Mrows / 128), 512, 0, stream>>>(A, Bw, bias, out,
                                                              K, mdiv, sq, sr);
  };
  auto lstm = [&](const float* xpv, const __bf16* Whh, const float* h0,
                  const float* c0, __bf16* outBf, float* outF, float* hT,
                  float* cT, int S_, int ldo, int co, int ldT, int rev) {
    lstm_seq<<<1, 512, 0, stream>>>(xpv, Whh, h0, c0, outBf, outF, hT, cT, S_,
                                    ldo, co, ldT, rev);
  };

  // Embeddings -> [T,B,E] bf16
  embed_kernel<<<dim3((unsigned)((M * kE + 255) / 256)), 256, 0, stream>>>(
      src_seq, src_emb, xsrc, kS);
  embed_kernel<<<dim3((unsigned)((M * kE + 255) / 256)), 256, 0, stream>>>(
      trg_seq, trg_emb, xtrg, kT);

  // Encoder layer 0 (fwd / bwd), outputs concat into enc0 [S,B,1024]
  gemm(xsrc, eW0f_b, eb0f, xp, 3072, 2048, 256, 1, 2048, 0);
  lstm(xp, eU0f_b, nullptr, nullptr, enc0, nullptr, hT0, cT0, kS, 1024, 0, 1024, 0);
  gemm(xsrc, eW0b_b, eb0b, xp, 3072, 2048, 256, 1, 2048, 0);
  lstm(xp, eU0b_b, nullptr, nullptr, enc0, nullptr, hT0, cT0, kS, 1024, 512, 1024, 1);

  // Encoder layer 1
  gemm(enc0, eW1f_b, eb1f, xp, 3072, 2048, 1024, 1, 2048, 0);
  lstm(xp, eU1f_b, nullptr, nullptr, enc1, nullptr, hT1, cT1, kS, 1024, 0, 1024, 0);
  gemm(enc0, eW1b_b, eb1b, xp, 3072, 2048, 1024, 1, 2048, 0);
  lstm(xp, eU1b_b, nullptr, nullptr, enc1, nullptr, hT1, cT1, kS, 1024, 512, 1024, 1);

  // enc_out = enc1 @ out_W^T + out_b  (fp32, [S,B,512])
  gemm(enc1, outW_b, out_b, encO, 3072, 512, 1024, 1, 512, 0);

  // Decoder initial states
  proj_kernel<<<64, 256, 0, stream>>>(hT0, hpW, hpb, h00);
  proj_kernel<<<64, 256, 0, stream>>>(hT1, hpW + 512 * 1024, hpb + 512, h01);
  proj_kernel<<<64, 256, 0, stream>>>(cT0, cpW, cpb, c00);
  proj_kernel<<<64, 256, 0, stream>>>(cT1, cpW + 512 * 1024, cpb + 512, c01);

  // Decoder layer 0 and 1
  gemm(xtrg, dW0_b, db0, xp, 3072, 2048, 256, 1, 2048, 0);
  lstm(xp, dU0_b, h00, c00, dec0, nullptr, nullptr, nullptr, kT, 512, 0, 0, 0);
  gemm(dec0, dW1_b, db1, xp, 3072, 2048, 512, 1, 2048, 0);
  lstm(xp, dU1_b, h01, c01, nullptr, yf, nullptr, nullptr, kT, 512, 0, 0, 0);

  // Attention -> combined bf16 [T,B,1024]
  attn_kernel<<<dim3(kT * kB), 128, 0, stream>>>(yf, encO, src_seq, comb);

  // logits[b,t,v]: m = t*32 + b -> out + t*V + b*(T*V)
  gemm(comb, fcW_b, fcb, logits, 3072, kV, 1024, kB, (long long)kV,
       (long long)kT * kV);
}
